// FPM_11020886082104
// MI455X (gfx1250) — matrix operand order repaired and checked by reading, not compiled, not run on hardware
//
#include <hip/hip_runtime.h>

// ---------------------------------------------------------------------------
// Types for CDNA5 WMMA (gfx1250, wave32)
// ---------------------------------------------------------------------------
typedef __attribute__((ext_vector_type(16))) __bf16          bf16x16;
typedef __attribute__((ext_vector_type(8)))  float           f32x8;
typedef __attribute__((ext_vector_type(8)))  unsigned int    u32x8;

#define Bsz   16
#define Nsz   96
#define Dsz   128
#define Lsz   95
#define ROWS  (Bsz * Nsz)      // 1536
#define GATE  (3 * Dsz)        // 384
#define WELEM (GATE * Dsz)     // 49152 elements per weight matrix
#define LDP   136              // padded LDS row stride (bf16): 68 dwords -> conflict-free
#define MFP   104              // padded stride for mfull tiles: 52 dwords -> conflict-free

// ---------------------------------------------------------------------------
// Helpers
// ---------------------------------------------------------------------------
__device__ __forceinline__ unsigned short f2bf(float f) {
    unsigned int u = __float_as_uint(f);
    unsigned int r = 0x7FFFu + ((u >> 16) & 1u);   // round-to-nearest-even
    return (unsigned short)((u + r) >> 16);
}

__device__ __forceinline__ float sigm(float x) { return 1.0f / (1.0f + __expf(-x)); }

__device__ __forceinline__ f32x8 wmma_bf16(bf16x16 a, bf16x16 b, f32x8 c) {
    return __builtin_amdgcn_wmma_f32_16x16x32_bf16(
        false, a, false, b, (short)0, c, false, false);
}

// A-matrix fragment, 16x32 bf16, from LDS (row-major, stride ld elems, ld % 8 == 0).
// K = h*16 + 2v (contiguous per lane half, matches B loaders).
__device__ __forceinline__ bf16x16 load_a_frag(const unsigned short* s, int ld, int k0) {
    int lane = threadIdx.x & 31;
    int m = lane & 15, h = lane >> 4;
    u32x8 a;
#pragma unroll
    for (int v = 0; v < 8; ++v) {
        int kk = k0 + h * 16 + 2 * v;
        a[v] = *(const unsigned int*)(s + m * ld + kk);
    }
    return __builtin_bit_cast(bf16x16, a);
}

// B-matrix fragment, 32x16 bf16, from global weights (row-major OxK):
// B[k][n] = W[(o0+n)*Dsz + k0+k]; per-lane 32B contiguous (2 x b128), L2-resident.
struct BF32B { uint4 lo, hi; };
__device__ __forceinline__ bf16x16 load_b_w(const unsigned short* w, int o0, int k0) {
    int lane = threadIdx.x & 31;
    int n = lane & 15, h = lane >> 4;
    const BF32B* p = reinterpret_cast<const BF32B*>(w + (o0 + n) * Dsz + k0 + h * 16);
    return __builtin_bit_cast(bf16x16, *p);
}

// B-matrix fragment from TRANSPOSED LDS tile (N-major x K, stride ld, ld % 8 == 0).
__device__ __forceinline__ bf16x16 load_b_ldsT(const unsigned short* s, int ld, int n0, int k0) {
    int lane = threadIdx.x & 31;
    int n = lane & 15, h = lane >> 4;
    const unsigned short* p = s + (n0 + n) * ld + k0 + h * 16;
    u32x8 b;
#pragma unroll
    for (int v = 0; v < 8; ++v) b[v] = *(const unsigned int*)(p + 2 * v);
    return __builtin_bit_cast(bf16x16, b);
}

// ---------------------------------------------------------------------------
// Kernel 1: weights f32 -> bf16 (once; ~384KB, stays L2-resident)
// ---------------------------------------------------------------------------
__global__ __launch_bounds__(256) void cvt_weights(
    const float* __restrict__ w0, const float* __restrict__ w1,
    const float* __restrict__ w2, const float* __restrict__ w3,
    unsigned short* __restrict__ out) {
    int i = blockIdx.x * 256 + threadIdx.x;
    int which = i / WELEM, j = i % WELEM;
    const float* src = (which == 0) ? w0 : (which == 1) ? w1 : (which == 2) ? w2 : w3;
    out[i] = f2bf(src[j]);
}

// ---------------------------------------------------------------------------
// Kernel 2: Mfull[b] = A[b] (96x96) @ cur[b] (96x128)  (leave-one-out base)
// ---------------------------------------------------------------------------
__global__ __launch_bounds__(256) void mfull_kernel(
    const float* __restrict__ A, const float* __restrict__ cur, float* __restrict__ M) {
    __shared__ __align__(16) unsigned short Ab[16 * MFP];
    __shared__ __align__(16) unsigned short Ct[Dsz * MFP];      // transposed (N-major)
    int b  = blockIdx.x / 6;
    int mt = blockIdx.x % 6;

    const float* As = A + (b * Nsz + mt * 16) * Nsz;
    for (int i = threadIdx.x; i < 16 * Nsz; i += 256)
        Ab[(i / Nsz) * MFP + (i % Nsz)] = f2bf(As[i]);
    const float* Cs = cur + b * Nsz * Dsz;
    for (int i = threadIdx.x; i < Nsz * Dsz; i += 256) {
        int k = i / Dsz, d = i % Dsz;
        Ct[d * MFP + k] = f2bf(Cs[i]);
    }
    __syncthreads();

    int w  = threadIdx.x >> 5;
    int d0 = w * 16;
    f32x8 acc = {};
#pragma unroll
    for (int kc = 0; kc < 3; ++kc) {
        bf16x16 a  = load_a_frag(Ab, MFP, kc * 32);
        bf16x16 bm = load_b_ldsT(Ct, MFP, d0, kc * 32);
        acc = wmma_bf16(a, bm, acc);
    }
    int lane = threadIdx.x & 31;
    int n = lane & 15, h = lane >> 4;
    float* Mo = M + (b * Nsz + mt * 16) * Dsz;
#pragma unroll
    for (int v = 0; v < 8; ++v) Mo[(v + 8 * h) * Dsz + d0 + n] = acc[v];
}

// ---------------------------------------------------------------------------
// Kernel 3: GI0[t][bn][384] = x_t @ W_ih0^T for ALL 95 timesteps at once.
// Fully parallel (9120 blocks) -- hoists 25% of the scan's critical path.
// ---------------------------------------------------------------------------
__global__ __launch_bounds__(256) void gi0_kernel(
    const float* __restrict__ A, const float* __restrict__ cur,
    const float* __restrict__ Mfull, const unsigned short* __restrict__ Wih0,
    float* __restrict__ GI0) {
    __shared__ __align__(16) unsigned short XT[16 * LDP];
    int t    = blockIdx.x / (ROWS / 16);
    int row0 = (blockIdx.x % (ROWS / 16)) * 16;

    {   // x_t = Mfull[loo] - A[loo,n]*cur[n]
        int r  = threadIdx.x >> 4;
        int d0 = (threadIdx.x & 15) * 8;
        int bn = row0 + r;
        int b = bn / Nsz, n = bn % Nsz;
        int rr = (t < n) ? t : t + 1;
        float aval = A[(b * Nsz + rr) * Nsz + n];
        const float* Mp = Mfull + (b * Nsz + rr) * Dsz;
        const float* Cp = cur   + (b * Nsz + n)  * Dsz;
#pragma unroll
        for (int q = 0; q < 8; ++q) {
            int d = d0 + q;
            XT[r * LDP + d] = f2bf(Mp[d] - aval * Cp[d]);
        }
    }
    __syncthreads();

    int w    = threadIdx.x >> 5;
    int lane = threadIdx.x & 31;
    int nl = lane & 15, hf = lane >> 4;
    int d0 = w * 16;

    f32x8 gr = {}, gz = {}, gn = {};
#pragma unroll
    for (int kc = 0; kc < 4; ++kc) {
        int k0 = kc * 32;
        bf16x16 ax = load_a_frag(XT, LDP, k0);
        gr = wmma_bf16(ax, load_b_w(Wih0, d0,           k0), gr);
        gz = wmma_bf16(ax, load_b_w(Wih0, Dsz + d0,     k0), gz);
        gn = wmma_bf16(ax, load_b_w(Wih0, 2 * Dsz + d0, k0), gn);
    }
    float* g = GI0 + ((size_t)t * ROWS) * GATE;
#pragma unroll
    for (int v = 0; v < 8; ++v) {
        size_t base = (size_t)(row0 + v + 8 * hf) * GATE;
        g[base + d0 + nl]           = gr[v];
        g[base + Dsz + d0 + nl]     = gz[v];
        g[base + 2 * Dsz + d0 + nl] = gn[v];
    }
}

// ---------------------------------------------------------------------------
// Kernel 4: PERSISTENT 95-step fused 2-layer GRU scan.
// The recurrence is row-local: each block owns 16 rows for the entire scan.
// h-states live in registers as C-fragments; the shared bf16 tile H0T doubles
// as layer1's A-operand (new h0 == layer1 input). 3 barriers per step.
// ---------------------------------------------------------------------------
__global__ __launch_bounds__(256) void gru_scan(
    const float* __restrict__ A, const float* __restrict__ cur,
    const float* __restrict__ Mfull, const float* __restrict__ GI0,
    const unsigned short* __restrict__ Wih0, const unsigned short* __restrict__ Whh0,
    const float* __restrict__ bih0, const float* __restrict__ bhh0,
    const unsigned short* __restrict__ Wih1, const unsigned short* __restrict__ Whh1,
    const float* __restrict__ bih1, const float* __restrict__ bhh1,
    float* __restrict__ out) {
    __shared__ __align__(16) unsigned short XT [16 * LDP];  // fallback only
    __shared__ __align__(16) unsigned short H0T[16 * LDP];  // h0 (bf16), == layer1 input
    __shared__ __align__(16) unsigned short H1T[16 * LDP];  // h1 (bf16)

    int row0 = blockIdx.x * 16;
    int w    = threadIdx.x >> 5;
    int lane = threadIdx.x & 31;
    int nl = lane & 15, hf = lane >> 4;
    int d0 = w * 16;
    int d  = d0 + nl;

    // ---- init: register h-fragments + bf16 tiles from cur ----
    f32x8 hp0, hp1;
#pragma unroll
    for (int v = 0; v < 8; ++v) {
        float c = cur[(size_t)(row0 + v + 8 * hf) * Dsz + d];
        hp0[v] = c; hp1[v] = c;
    }
    {
        int r   = threadIdx.x >> 4;
        int dd0 = (threadIdx.x & 15) * 8;
        const float* Cp = cur + (size_t)(row0 + r) * Dsz;
#pragma unroll
        for (int q = 0; q < 8; ++q) {
            unsigned short bv = f2bf(Cp[dd0 + q]);
            H0T[r * LDP + dd0 + q] = bv;
            H1T[r * LDP + dd0 + q] = bv;
        }
    }
    // per-wave bias scalars (registers for the whole scan)
    float bir0 = bih0[d], biz0 = bih0[Dsz + d], bin0 = bih0[2 * Dsz + d];
    float bhr0 = bhh0[d], bhz0 = bhh0[Dsz + d], bhn0 = bhh0[2 * Dsz + d];
    float bir1 = bih1[d], biz1 = bih1[Dsz + d], bin1 = bih1[2 * Dsz + d];
    float bhr1 = bhh1[d], bhz1 = bhh1[Dsz + d], bhn1 = bhh1[2 * Dsz + d];
    __syncthreads();

    for (int t = 0; t < Lsz; ++t) {
        // ---- optional x_t staging (only when GI0 was not precomputed) ----
        if (!GI0) {
            int r   = threadIdx.x >> 4;
            int dd0 = (threadIdx.x & 15) * 8;
            int bn = row0 + r;
            int b = bn / Nsz, n = bn % Nsz;
            int rr = (t < n) ? t : t + 1;
            float aval = A[(b * Nsz + rr) * Nsz + n];
            const float* Mp = Mfull + (b * Nsz + rr) * Dsz;
            const float* Cp = cur   + (b * Nsz + n)  * Dsz;
#pragma unroll
            for (int q = 0; q < 8; ++q)
                XT[r * LDP + dd0 + q] = f2bf(Mp[dd0 + q] - aval * Cp[dd0 + q]);
            __syncthreads();
        }

        // ---- layer 0 matmuls ----
        f32x8 ir = {}, iz = {}, in_ = {}, hr = {}, hz = {}, hn = {};
        if (GI0) {
            const float* g = GI0 + ((size_t)t * ROWS) * GATE;
#pragma unroll
            for (int v = 0; v < 8; ++v) {
                size_t base = (size_t)(row0 + v + 8 * hf) * GATE;
                ir[v]  = g[base + d];
                iz[v]  = g[base + Dsz + d];
                in_[v] = g[base + 2 * Dsz + d];
            }
        }
#pragma unroll
        for (int kc = 0; kc < 4; ++kc) {
            int k0 = kc * 32;
            if (!GI0) {
                bf16x16 ax = load_a_frag(XT, LDP, k0);
                ir  = wmma_bf16(ax, load_b_w(Wih0, d0,           k0), ir);
                iz  = wmma_bf16(ax, load_b_w(Wih0, Dsz + d0,     k0), iz);
                in_ = wmma_bf16(ax, load_b_w(Wih0, 2 * Dsz + d0, k0), in_);
            }
            bf16x16 ah = load_a_frag(H0T, LDP, k0);
            hr = wmma_bf16(ah, load_b_w(Whh0, d0,           k0), hr);
            hz = wmma_bf16(ah, load_b_w(Whh0, Dsz + d0,     k0), hz);
            hn = wmma_bf16(ah, load_b_w(Whh0, 2 * Dsz + d0, k0), hn);
        }
        __syncthreads();                       // all waves done reading old H0T / XT

        // ---- layer 0 gates: update register state, publish new h0 tile ----
#pragma unroll
        for (int v = 0; v < 8; ++v) {
            float rg = sigm(ir[v] + bir0 + hr[v] + bhr0);
            float zg = sigm(iz[v] + biz0 + hz[v] + bhz0);
            float ng = tanhf(in_[v] + bin0 + rg * (hn[v] + bhn0));
            float hv = (1.0f - zg) * ng + zg * hp0[v];
            hp0[v] = hv;
            H0T[(v + 8 * hf) * LDP + d] = f2bf(hv);   // own (row,col): no race
        }
        __syncthreads();                       // new H0T visible to all waves

        // ---- layer 1 matmuls: gi from new h0, gh from old h1 ----
        f32x8 jr = {}, jz = {}, jn = {}, kr = {}, kz = {}, kn = {};
#pragma unroll
        for (int kc = 0; kc < 4; ++kc) {
            int k0 = kc * 32;
            bf16x16 ax = load_a_frag(H0T, LDP, k0);
            bf16x16 ah = load_a_frag(H1T, LDP, k0);
            jr = wmma_bf16(ax, load_b_w(Wih1, d0,           k0), jr);
            jz = wmma_bf16(ax, load_b_w(Wih1, Dsz + d0,     k0), jz);
            jn = wmma_bf16(ax, load_b_w(Wih1, 2 * Dsz + d0, k0), jn);
            kr = wmma_bf16(ah, load_b_w(Whh1, d0,           k0), kr);
            kz = wmma_bf16(ah, load_b_w(Whh1, Dsz + d0,     k0), kz);
            kn = wmma_bf16(ah, load_b_w(Whh1, 2 * Dsz + d0, k0), kn);
        }
        __syncthreads();                       // all waves done reading old H1T

        // ---- layer 1 gates ----
#pragma unroll
        for (int v = 0; v < 8; ++v) {
            float rg = sigm(jr[v] + bir1 + kr[v] + bhr1);
            float zg = sigm(jz[v] + biz1 + kz[v] + bhz1);
            float ng = tanhf(jn[v] + bin1 + rg * (kn[v] + bhn1));
            float hv = (1.0f - zg) * ng + zg * hp1[v];
            hp1[v] = hv;
            H1T[(v + 8 * hf) * LDP + d] = f2bf(hv);
        }
        // next read of H1T is after next step's two barriers -> safe
    }

    // ---- write final h1 (block owns these rows exclusively) ----
#pragma unroll
    for (int v = 0; v < 8; ++v)
        out[(size_t)(row0 + v + 8 * hf) * Dsz + d] = hp1[v];
}

// ---------------------------------------------------------------------------
// Launch: ~8 graph nodes total (no per-timestep launches)
// ---------------------------------------------------------------------------
extern "C" void kernel_launch(void* const* d_in, const int* in_sizes, int n_in,
                              void* d_out, int out_size, void* d_ws, size_t ws_size,
                              hipStream_t stream) {
    (void)in_sizes; (void)n_in; (void)out_size;
    const float* A     = (const float*)d_in[0];
    const float* emb   = (const float*)d_in[1];
    const float* Wih0f = (const float*)d_in[2];
    const float* Whh0f = (const float*)d_in[3];
    const float* bih0  = (const float*)d_in[4];
    const float* bhh0  = (const float*)d_in[5];
    const float* Wih1f = (const float*)d_in[6];
    const float* Whh1f = (const float*)d_in[7];
    const float* bih1  = (const float*)d_in[8];
    const float* bhh1  = (const float*)d_in[9];

    const size_t NELT = (size_t)ROWS * Dsz;                    // 196608
    const size_t BASE = 4 * WELEM * 2 + 2 * NELT * 4;          // weights + cur + Mfull
    const size_t GI0B = (size_t)Lsz * ROWS * GATE * 4;         // ~214 MiB

    unsigned short* Wih0 = (unsigned short*)d_ws;
    unsigned short* Whh0 = Wih0 + WELEM;
    unsigned short* Wih1 = Whh0 + WELEM;
    unsigned short* Whh1 = Wih1 + WELEM;
    float* cur   = (float*)(Whh1 + WELEM);
    float* Mfull = cur   + NELT;
    float* GI0   = Mfull + NELT;
    const bool use_gi0 = (ws_size >= BASE + GI0B);
    const float* gi0p = use_gi0 ? GI0 : nullptr;

    cvt_weights<<<(4 * WELEM) / 256, 256, 0, stream>>>(Wih0f, Whh0f, Wih1f, Whh1f, Wih0);
    hipMemcpyAsync(cur, emb, NELT * sizeof(float), hipMemcpyDeviceToDevice, stream);

    for (int iter = 0; iter < 2; ++iter) {
        mfull_kernel<<<Bsz * 6, 256, 0, stream>>>(A, cur, Mfull);
        if (use_gi0)
            gi0_kernel<<<Lsz * (ROWS / 16), 256, 0, stream>>>(A, cur, Mfull, Wih0, GI0);
        float* out = (iter == 1) ? (float*)d_out : cur;   // in-place safe: rows are block-exclusive
        gru_scan<<<ROWS / 16, 256, 0, stream>>>(A, cur, Mfull, gi0p,
                                                Wih0, Whh0, bih0, bhh0,
                                                Wih1, Whh1, bih1, bhh1, out);
    }
}
